// GATLayer_36421322670606
// MI455X (gfx1250) — matrix operand order repaired and checked by reading, not compiled, not run on hardware
//
#include <hip/hip_runtime.h>

#define GAT_N 4096
#define GAT_K 256     // IN_DIM
#define GAT_O 256     // H*D
#define GAT_H 4
#define GAT_D 64
#define GAT_ALPHA 0.2f

typedef __attribute__((ext_vector_type(16))) _Float16 v16h;
typedef __attribute__((ext_vector_type(8)))  float    v8f;

__device__ __forceinline__ v16h pack16(float4 x0, float4 x1, float4 x2, float4 x3) {
  v16h r;
  r[0] = (_Float16)x0.x;  r[1] = (_Float16)x0.y;  r[2] = (_Float16)x0.z;  r[3] = (_Float16)x0.w;
  r[4] = (_Float16)x1.x;  r[5] = (_Float16)x1.y;  r[6] = (_Float16)x1.z;  r[7] = (_Float16)x1.w;
  r[8] = (_Float16)x2.x;  r[9] = (_Float16)x2.y;  r[10] = (_Float16)x2.z; r[11] = (_Float16)x2.w;
  r[12] = (_Float16)x3.x; r[13] = (_Float16)x3.y; r[14] = (_Float16)x3.z; r[15] = (_Float16)x3.w;
  return r;
}

// ---------------------------------------------------------------------------
// Kernel A: Wh = h @ W^T + b  (f16 WMMA, f32 accumulate)
// grid = (N/16, O/16), block = 32 (one wave per 16x16 tile)
// Writes Whf32 [N][O] (for src/dst dots) and WhT16 [O][N] (B-frag friendly).
// ---------------------------------------------------------------------------
__global__ __launch_bounds__(32)
void gat_gemm_wh(const float* __restrict__ hin, const float* __restrict__ W,
                 const float* __restrict__ bias,
                 float* __restrict__ Whf32, _Float16* __restrict__ WhT16)
{
  const int lane = threadIdx.x & 31;
  const int r    = lane & 15;
  const int hi   = lane >> 4;
  const int rowbase = blockIdx.x * 16;
  const int colbase = blockIdx.y * 16;
  const int col  = colbase + r;

  const float* hrow = hin + (size_t)(rowbase + r) * GAT_K;
  const float* wrow = W   + (size_t)col * GAT_K;

  v8f acc = {};
  #pragma unroll
  for (int kb = 0; kb < GAT_K; kb += 32) {
    // A frag: row = lane%16, K in {hi*8+0..7} U {hi*8+16..23}
    const float4* ap0 = (const float4*)(hrow + kb + hi * 8);
    const float4* ap1 = (const float4*)(hrow + kb + hi * 8 + 16);
    v16h af = pack16(ap0[0], ap0[1], ap1[0], ap1[1]);
    // B frag: col = lane%16, K in {hi*8+0..7} U {hi*8+16..23} (B[k][n] = W[n][k])
    const float4* bp0 = (const float4*)(wrow + kb + hi * 8);
    const float4* bp1 = (const float4*)(wrow + kb + hi * 8 + 16);
    v16h bf = pack16(bp0[0], bp0[1], bp1[0], bp1[1]);
    acc = __builtin_amdgcn_wmma_f32_16x16x32_f16(false, af, false, bf,
                                                 (short)0, acc, false, false);
  }
  const float bv = bias[col];
  #pragma unroll
  for (int k = 0; k < 8; ++k) {
    const int m = k + hi * 8;
    const float v = acc[k] + bv;
    Whf32[(size_t)(rowbase + m) * GAT_O + col] = v;
    WhT16[(size_t)col * GAT_N + rowbase + m]   = (_Float16)v;
  }
}

// ---------------------------------------------------------------------------
// Kernel B: src/dst attention dots (transposed layout) + per-head global
// max of dst (LDS tree reduction, single block -> no atomics, no state).
// ---------------------------------------------------------------------------
__global__ __launch_bounds__(1024)
void gat_srcdst(const float* __restrict__ Whf32, const float* __restrict__ avec,
                float* __restrict__ srcT, float* __restrict__ dstT,
                float* __restrict__ dmax)
{
  __shared__ float red[GAT_H][1024];
  const int t = threadIdx.x;
  float lmax[GAT_H];
  #pragma unroll
  for (int hh = 0; hh < GAT_H; ++hh) lmax[hh] = -3.0e38f;

  for (int i = t; i < GAT_N; i += 1024) {
    const float* wr = Whf32 + (size_t)i * GAT_O;
    #pragma unroll
    for (int hh = 0; hh < GAT_H; ++hh) {
      const float* av = avec + hh * (2 * GAT_D);
      float s = 0.f, d = 0.f;
      #pragma unroll 8
      for (int dd = 0; dd < GAT_D; ++dd) {
        const float w = wr[hh * GAT_D + dd];
        s += w * av[dd];
        d += w * av[GAT_D + dd];
      }
      srcT[hh * GAT_N + i] = s;
      dstT[hh * GAT_N + i] = d;
      lmax[hh] = fmaxf(lmax[hh], d);
    }
  }
  #pragma unroll
  for (int hh = 0; hh < GAT_H; ++hh) red[hh][t] = lmax[hh];
  __syncthreads();
  for (int sft = 512; sft > 0; sft >>= 1) {
    if (t < sft) {
      #pragma unroll
      for (int hh = 0; hh < GAT_H; ++hh)
        red[hh][t] = fmaxf(red[hh][t], red[hh][t + sft]);
    }
    __syncthreads();
  }
  if (t < GAT_H) dmax[t] = red[t][0];
}

// ---------------------------------------------------------------------------
// Kernel C: fused masked-softmax attention + aggregation.
// grid = N/16 blocks, block = 128 (wave w == head w, 16 output rows).
// One pass over j: p~ = adj ? exp(leakyrelu(src_i+dst_j) - m_i) : 0, packed
// straight into the 16x32 A fragment; 4 WMMAs (D = 4 x 16 cols); row sums
// accumulated per lane and applied to the accumulators at the end.
// m_i = leakyrelu(src_i + max_j dst_j) upper-bounds every masked score.
// ---------------------------------------------------------------------------
__global__ __launch_bounds__(128)
void gat_attn(const int* __restrict__ adj, const _Float16* __restrict__ WhT16,
              const float* __restrict__ srcT, const float* __restrict__ dstT,
              const float* __restrict__ dmax, float* __restrict__ out)
{
  const int wave = threadIdx.x >> 5;     // head index
  const int lane = threadIdx.x & 31;
  const int r    = lane & 15;
  const int hi   = lane >> 4;
  const int rowbase = blockIdx.x * 16;
  const int row  = rowbase + r;
  const int h    = wave;

  const float srcv = srcT[h * GAT_N + row];
  const float tmx  = srcv + dmax[h];
  const float mrow = tmx > 0.f ? tmx : GAT_ALPHA * tmx;

  const int*      adjrow = adj  + (size_t)row * GAT_N;
  const float*    dsth   = dstT + (size_t)h * GAT_N;
  const _Float16* b0p = WhT16 + (size_t)(h * GAT_D +  0 + r) * GAT_N;
  const _Float16* b1p = WhT16 + (size_t)(h * GAT_D + 16 + r) * GAT_N;
  const _Float16* b2p = WhT16 + (size_t)(h * GAT_D + 32 + r) * GAT_N;
  const _Float16* b3p = WhT16 + (size_t)(h * GAT_D + 48 + r) * GAT_N;

  v8f acc0 = {}, acc1 = {}, acc2 = {}, acc3 = {};
  float lanesum = 0.f;

  for (int jb = 0; jb < GAT_N; jb += 32) {
    const int k0 = jb + hi * 16;
    const int4* ap  = (const int4*)(adjrow + k0);
    const int4* ap2 = (const int4*)(adjrow + k0 + 8);
    const int4 a0 = ap[0],  a1 = ap[1];
    const int4 a2 = ap2[0], a3 = ap2[1];
    const float4* dp  = (const float4*)(dsth + k0);
    const float4* dp2 = (const float4*)(dsth + k0 + 8);
    const float4 d0 = dp[0],  d1 = dp[1];
    const float4 d2 = dp2[0], d3 = dp2[1];

    v16h af;
#define GAT_EDGE(i, A, Dv) { float e_ = srcv + (Dv);                         \
    e_ = e_ > 0.f ? e_ : GAT_ALPHA * e_;                                     \
    float p_ = ((A) == 1) ? __expf(e_ - mrow) : 0.f;                         \
    lanesum += p_; af[i] = (_Float16)p_; }
    GAT_EDGE(0,  a0.x, d0.x) GAT_EDGE(1,  a0.y, d0.y)
    GAT_EDGE(2,  a0.z, d0.z) GAT_EDGE(3,  a0.w, d0.w)
    GAT_EDGE(4,  a1.x, d1.x) GAT_EDGE(5,  a1.y, d1.y)
    GAT_EDGE(6,  a1.z, d1.z) GAT_EDGE(7,  a1.w, d1.w)
    GAT_EDGE(8,  a2.x, d2.x) GAT_EDGE(9,  a2.y, d2.y)
    GAT_EDGE(10, a2.z, d2.z) GAT_EDGE(11, a2.w, d2.w)
    GAT_EDGE(12, a3.x, d3.x) GAT_EDGE(13, a3.y, d3.y)
    GAT_EDGE(14, a3.z, d3.z) GAT_EDGE(15, a3.w, d3.w)
#undef GAT_EDGE

    const int kb2 = jb + hi * 16;
    const v16h bf0 = *(const v16h*)(b0p + kb2);
    const v16h bf1 = *(const v16h*)(b1p + kb2);
    const v16h bf2 = *(const v16h*)(b2p + kb2);
    const v16h bf3 = *(const v16h*)(b3p + kb2);
    acc0 = __builtin_amdgcn_wmma_f32_16x16x32_f16(false, af, false, bf0, (short)0, acc0, false, false);
    acc1 = __builtin_amdgcn_wmma_f32_16x16x32_f16(false, af, false, bf1, (short)0, acc1, false, false);
    acc2 = __builtin_amdgcn_wmma_f32_16x16x32_f16(false, af, false, bf2, (short)0, acc2, false, false);
    acc3 = __builtin_amdgcn_wmma_f32_16x16x32_f16(false, af, false, bf3, (short)0, acc3, false, false);
  }

  // Row sums: lanes l and l+16 hold disjoint j-subsets of row r = l%16.
  const float tot = lanesum + __shfl_xor(lanesum, 16, 32);
  __shared__ float wsum[GAT_H][16];
  if (lane < 16) wsum[wave][r] = tot;
  __syncthreads();

  #pragma unroll
  for (int k = 0; k < 8; ++k) {
    const int m = k + hi * 8;             // C row for this accumulator slot
    float s = wsum[h][m];
    s = (s > 0.f) ? s : 1.f;              // isolated-node guard
    const float inv = 1.f / s;
    float* op = out + (size_t)(rowbase + m) * GAT_O + h * GAT_D;
    op[ 0 + r] = acc0[k] * inv;
    op[16 + r] = acc1[k] * inv;
    op[32 + r] = acc2[k] * inv;
    op[48 + r] = acc3[k] * inv;
  }
}

// ---------------------------------------------------------------------------
extern "C" void kernel_launch(void* const* d_in, const int* in_sizes, int n_in,
                              void* d_out, int out_size, void* d_ws, size_t ws_size,
                              hipStream_t stream) {
  const float* hin  = (const float*)d_in[0];   // [N, IN_DIM] f32
  const int*   adj  = (const int*)  d_in[1];   // [N, N] int32
  const float* W    = (const float*)d_in[2];   // [O, IN_DIM] f32
  const float* bias = (const float*)d_in[3];   // [O] f32
  const float* avec = (const float*)d_in[4];   // [H, 2D] f32

  char* ws = (char*)d_ws;
  float*     Whf32 = (float*)ws;                                 // 4 MB
  _Float16*  WhT16 = (_Float16*)(ws + (4u << 20));               // 2 MB
  float*     srcT  = (float*)(ws + (6u << 20));                  // 64 KB
  float*     dstT  = (float*)(ws + (6u << 20) + (64u << 10));    // 64 KB
  float*     dmax  = (float*)(ws + (6u << 20) + (128u << 10));   // 16 B

  dim3 gA(GAT_N / 16, GAT_O / 16);
  hipLaunchKernelGGL(gat_gemm_wh, gA, dim3(32), 0, stream,
                     hin, W, bias, Whf32, WhT16);
  hipLaunchKernelGGL(gat_srcdst, dim3(1), dim3(1024), 0, stream,
                     Whf32, avec, srcT, dstT, dmax);
  hipLaunchKernelGGL(gat_attn, dim3(GAT_N / 16), dim3(128), 0, stream,
                     adj, WhT16, srcT, dstT, dmax, (float*)d_out);
}
